// LocalEncoder_3659312136618
// MI455X (gfx1250) — matrix operand order repaired and checked by reading, not compiled, not run on hardware
//
#include <hip/hip_runtime.h>
#include <math.h>

typedef __attribute__((ext_vector_type(16))) _Float16 v16h;
typedef __attribute__((ext_vector_type(8)))  float    v8f;

namespace cfg {
constexpr int B   = 2;
constexpr int S   = 2048;
constexpr int D   = 768;
constexpr int H   = 12;
constexpr int HD  = 64;          // D / H
constexpr int FF  = 3072;        // 4*D
constexpr int KQ  = 256;         // patches per sequence
constexpr int BKk = 2;           // blt_k
constexpr int DG  = 2048;        // d_global
constexpr int HV  = 20000;       // hash vocab
constexpr int NG  = 6;           // group sizes 3..8
constexpr int W   = 512;         // sliding window
constexpr int BS  = B * S;
constexpr float SCALE = 0.125f;  // 1/sqrt(64)
}

// ---------------------------------------------------------------------------
// 1) byte embedding + multi-group rolling-hash embeddings
// ---------------------------------------------------------------------------
__global__ void embed_hash_kernel(const int* __restrict__ tok,
                                  const float* __restrict__ emb,
                                  const float* __restrict__ ht,
                                  float* __restrict__ h)
{
    using namespace cfg;
    __shared__ int hashes[NG];
    const int bs = blockIdx.x;           // 0..B*S-1
    const int b  = bs / S;
    const int s  = bs % S;
    if (threadIdx.x < NG) {
        const int g = 3 + threadIdx.x;   // group sizes 3..8
        long long acc = 0, m = 1;
        for (int j = 0; j < g; ++j) {
            const int sp = s - j;
            const long long tv = (sp >= 0) ? (long long)tok[b * S + sp] : 0ll;
            acc = (acc + tv * (m % HV)) % HV;
            m *= 31;
        }
        hashes[threadIdx.x] = (int)acc;
    }
    __syncthreads();
    const int t0 = tok[bs];
    for (int d = threadIdx.x; d < D; d += blockDim.x) {
        float v = emb[(size_t)t0 * D + d];
        #pragma unroll
        for (int i = 0; i < NG; ++i)
            v += ht[((size_t)i * HV + hashes[i]) * D + d];
        h[(size_t)bs * D + d] = v;
    }
}

// ---------------------------------------------------------------------------
// 2) RMSNorm (safe for in-place y==x)
// ---------------------------------------------------------------------------
__global__ void rmsnorm_kernel(const float* x, const float* __restrict__ sc,
                               float* y, int ncol)
{
    __shared__ float red[256];
    const int row = blockIdx.x;
    const float* xr = x + (size_t)row * ncol;
    float ss = 0.f;
    for (int c = threadIdx.x; c < ncol; c += 256) { const float v = xr[c]; ss += v * v; }
    red[threadIdx.x] = ss; __syncthreads();
    for (int off = 128; off > 0; off >>= 1) {
        if (threadIdx.x < off) red[threadIdx.x] += red[threadIdx.x + off];
        __syncthreads();
    }
    const float inv = rsqrtf(red[0] / (float)ncol + 1e-5f);
    for (int c = threadIdx.x; c < ncol; c += 256)
        y[(size_t)row * ncol + c] = xr[c] * inv * sc[c];
}

// ---------------------------------------------------------------------------
// 3a) Pack A (MxK f32 row-major) -> f16 A-fragments [mt][kt][lane][e]
//     A-fragment element mapping (ISA 7.12.2): k = e + 8*((e>>3)+half), m = l16
// ---------------------------------------------------------------------------
__global__ __launch_bounds__(32)
void pack_a_kernel(const float* __restrict__ A, _Float16* __restrict__ P,
                   int M, int K)
{
    const int lane = threadIdx.x & 31;
    const int half = lane >> 4;
    const int l16  = lane & 15;
    const int mt = blockIdx.x, kt = blockIdx.y;
    const int KT = K >> 5;
    const int row = mt * 16 + l16;
    _Float16* dst = P + ((size_t)(mt * KT + kt) * 32 + lane) * 16;
    #pragma unroll
    for (int e = 0; e < 16; ++e) {
        const int kk = kt * 32 + e + 8 * ((e >> 3) + half);
        dst[e] = (_Float16)A[(size_t)row * K + kk];
    }
}

// ---------------------------------------------------------------------------
// 3b) Pack B (KxN f32 row-major) -> f16 B-fragments [nt][kt][lane][e]
//     B-fragment element mapping: k = e + 8*((e>>3)+half), n = l16
// ---------------------------------------------------------------------------
__global__ __launch_bounds__(32)
void pack_b_kernel(const float* __restrict__ Bw, _Float16* __restrict__ P,
                   int K, int N)
{
    const int lane = threadIdx.x & 31;
    const int half = lane >> 4;
    const int l16  = lane & 15;
    const int nt = blockIdx.x, kt = blockIdx.y;
    const int KT = K >> 5;
    const int n = nt * 16 + l16;
    _Float16* dst = P + ((size_t)(nt * KT + kt) * 32 + lane) * 16;
    #pragma unroll
    for (int e = 0; e < 16; ++e) {
        const int kk = kt * 32 + e + 8 * ((e >> 3) + half);
        dst[e] = (_Float16)Bw[(size_t)kk * N + n];
    }
}

// ---------------------------------------------------------------------------
// 3c) Packed WMMA GEMM: C[M,N] = op(A@B (+bias)); one wave owns a 32x64 tile
//     (2 A-frags x 4 B-frags -> 8 v_wmma per k-step, all operands loaded as
//      aligned v16h = coalesced b128 pairs).
//     flags bit0: accumulate into existing C; bit1: SiLU epilogue
// ---------------------------------------------------------------------------
__global__ __launch_bounds__(32)
void wmma_gemm_packed_kernel(const _Float16* __restrict__ pA,
                             const _Float16* __restrict__ pB,
                             const float* __restrict__ bias, float* C,
                             int M, int N, int K, int flags)
{
    const int lane = threadIdx.x & 31;
    const int half = lane >> 4;
    const int l16  = lane & 15;
    const int KT   = K >> 5;
    const int mt0  = blockIdx.x * 2;   // two 16-row tiles
    const int nt0  = blockIdx.y * 4;   // four 16-col tiles

    v8f acc[2][4];
    #pragma unroll
    for (int r = 0; r < 2; ++r)
        #pragma unroll
        for (int t = 0; t < 4; ++t) {
            if (flags & 1) {
                const int m0 = (mt0 + r) * 16;
                const int n  = (nt0 + t) * 16 + l16;
                #pragma unroll
                for (int v = 0; v < 8; ++v)
                    acc[r][t][v] = C[(size_t)(m0 + v + 8 * half) * N + n];
            } else {
                acc[r][t] = (v8f){0.f,0.f,0.f,0.f,0.f,0.f,0.f,0.f};
            }
        }

    const v16h* fA = (const v16h*)pA;
    const v16h* fB = (const v16h*)pB;
    for (int kt = 0; kt < KT; ++kt) {
        if (kt + 1 < KT) {   // cover load-use latency of the next k-step
            __builtin_prefetch(&fA[((size_t)mt0 * KT + kt + 1) * 32 + lane], 0, 3);
            __builtin_prefetch(&fB[((size_t)nt0 * KT + kt + 1) * 32 + lane], 0, 3);
        }
        const v16h a0 = fA[((size_t)(mt0 + 0) * KT + kt) * 32 + lane];
        const v16h a1 = fA[((size_t)(mt0 + 1) * KT + kt) * 32 + lane];
        #pragma unroll
        for (int t = 0; t < 4; ++t) {
            const v16h bf = fB[((size_t)(nt0 + t) * KT + kt) * 32 + lane];
            acc[0][t] = __builtin_amdgcn_wmma_f32_16x16x32_f16(
                false, a0, false, bf, (short)0, acc[0][t], false, false);
            acc[1][t] = __builtin_amdgcn_wmma_f32_16x16x32_f16(
                false, a1, false, bf, (short)0, acc[1][t], false, false);
        }
    }

    #pragma unroll
    for (int r = 0; r < 2; ++r) {
        const int m0 = (mt0 + r) * 16;
        #pragma unroll
        for (int t = 0; t < 4; ++t) {
            const int n = (nt0 + t) * 16 + l16;
            const float bv = bias ? bias[n] : 0.f;
            #pragma unroll
            for (int v = 0; v < 8; ++v) {
                float val = acc[r][t][v] + bv;
                if (flags & 2) val = val / (1.f + __expf(-val));   // SiLU
                C[(size_t)(m0 + v + 8 * half) * N + n] = val;
            }
        }
    }
}

// ---------------------------------------------------------------------------
// 4a) Pack attention K into score B-fragments: [b][h][jt(16 keys)][c(hd chunk)]
// ---------------------------------------------------------------------------
__global__ __launch_bounds__(32)
void pack_attn_k_kernel(const float* __restrict__ Kb, _Float16* __restrict__ P)
{
    using namespace cfg;
    const int lane = threadIdx.x & 31;
    const int half = lane >> 4;
    const int l16  = lane & 15;
    const int jt = blockIdx.x;   // S/16
    const int hh = blockIdx.y;
    const int bb = blockIdx.z;
    #pragma unroll
    for (int c = 0; c < 2; ++c) {
        _Float16* dst = P + ((((size_t)(bb * H + hh) * (S / 16) + jt) * 2 + c) * 32 + lane) * 16;
        #pragma unroll
        for (int e = 0; e < 16; ++e) {
            const int kk = c * 32 + e + 8 * ((e >> 3) + half);
            dst[e] = (_Float16)Kb[((size_t)(bb * S + jt * 16 + l16)) * D + hh * HD + kk];
        }
    }
}

// ---------------------------------------------------------------------------
// 4b) Pack attention V into PV B-fragments: [b][h][jb(32 keys)][t(hd col tile)]
// ---------------------------------------------------------------------------
__global__ __launch_bounds__(32)
void pack_attn_v_kernel(const float* __restrict__ Vb, _Float16* __restrict__ P)
{
    using namespace cfg;
    const int lane = threadIdx.x & 31;
    const int half = lane >> 4;
    const int l16  = lane & 15;
    const int jb = blockIdx.x;   // S/32
    const int hh = blockIdx.y;
    const int bb = blockIdx.z;
    #pragma unroll
    for (int t = 0; t < 4; ++t) {
        _Float16* dst = P + ((((size_t)(bb * H + hh) * (S / 32) + jb) * 4 + t) * 32 + lane) * 16;
        #pragma unroll
        for (int e = 0; e < 16; ++e) {
            const int jr = jb * 32 + e + 8 * ((e >> 3) + half);
            dst[e] = (_Float16)Vb[((size_t)(bb * S + jr)) * D + hh * HD + t * 16 + l16];
        }
    }
}

// ---------------------------------------------------------------------------
// 4c) Sliding-window causal attention, flash style with WMMA; K/V pre-packed.
// ---------------------------------------------------------------------------
__global__ __launch_bounds__(32)
void attn_sw_kernel(const float* __restrict__ Qb, const _Float16* __restrict__ fKp,
                    const _Float16* __restrict__ fVp, float* __restrict__ Ob)
{
    using namespace cfg;
    __shared__ float pls[16 * 32];
    const int lane = threadIdx.x & 31;
    const int half = lane >> 4;
    const int l16  = lane & 15;
    const int i0   = blockIdx.x * 16;   // query tile start
    const int hh   = blockIdx.y;
    const int bb   = blockIdx.z;
    const v16h* fK = (const v16h*)fKp;
    const v16h* fV = (const v16h*)fVp;
    const size_t kbase = (size_t)(bb * H + hh) * (S / 16) * 2;
    const size_t vbase = (size_t)(bb * H + hh) * (S / 32) * 4;

    // load Q fragments once (hd=64 -> two A-fragments of k=32)
    v16h qa[2];
    #pragma unroll
    for (int c = 0; c < 2; ++c)
        #pragma unroll
        for (int e = 0; e < 16; ++e) {
            const int kk = c * 32 + e + 8 * ((e >> 3) + half);
            qa[c][e] = (_Float16)Qb[((size_t)(bb * S + i0 + l16)) * D + hh * HD + kk];
        }

    v8f o[4];
    #pragma unroll
    for (int t = 0; t < 4; ++t) o[t] = (v8f){0.f,0.f,0.f,0.f,0.f,0.f,0.f,0.f};
    float mrow[8], lrow[8];
    #pragma unroll
    for (int v = 0; v < 8; ++v) { mrow[v] = -1e30f; lrow[v] = 0.f; }

    int jlo = i0 - (W - 1); if (jlo < 0) jlo = 0;
    jlo &= ~31;
    for (int j0 = jlo; j0 <= i0 + 15; j0 += 32) {
        // scores for 32 keys = two 16x16 D-fragments
        v8f s01[2];
        #pragma unroll
        for (int g = 0; g < 2; ++g) {
            v8f sc = (v8f){0.f,0.f,0.f,0.f,0.f,0.f,0.f,0.f};
            const int jt = (j0 >> 4) + g;
            #pragma unroll
            for (int c = 0; c < 2; ++c) {
                const v16h kf = fK[(kbase + (size_t)jt * 2 + c) * 32 + lane];
                sc = __builtin_amdgcn_wmma_f32_16x16x32_f16(
                    false, qa[c], false, kf, (short)0, sc, false, false);
            }
            s01[g] = sc;
        }
        // mask + online softmax (D-fragment row v: query row = i0+v+8*half)
        float alpha[8];
        #pragma unroll
        for (int v = 0; v < 8; ++v) {
            const int irow = i0 + v + 8 * half;
            float best = -1e30f;
            #pragma unroll
            for (int g = 0; g < 2; ++g) {
                const int jc = j0 + g * 16 + l16;
                const bool valid = (jc <= irow) && (irow - jc < W);
                const float sv = valid ? s01[g][v] * SCALE : -1e30f;
                s01[g][v] = sv;
                best = fmaxf(best, sv);
            }
            #pragma unroll
            for (int d = 1; d < 16; d <<= 1)
                best = fmaxf(best, __shfl_xor(best, d, 32));
            const float nm = fmaxf(mrow[v], best);
            alpha[v] = __expf(mrow[v] - nm);
            float rs = 0.f;
            #pragma unroll
            for (int g = 0; g < 2; ++g) {
                const float sv = s01[g][v];
                const float p = (sv <= -1e29f) ? 0.f : __expf(sv - nm);
                s01[g][v] = p;
                rs += p;
            }
            #pragma unroll
            for (int d = 1; d < 16; d <<= 1)
                rs += __shfl_xor(rs, d, 32);
            lrow[v] = lrow[v] * alpha[v] + rs;
            mrow[v] = nm;
        }
        #pragma unroll
        for (int t = 0; t < 4; ++t)
            #pragma unroll
            for (int v = 0; v < 8; ++v)
                o[t][v] *= alpha[v];
        // transpose P (D-fragment -> A-fragment) via LDS
        __syncthreads();
        #pragma unroll
        for (int g = 0; g < 2; ++g)
            #pragma unroll
            for (int v = 0; v < 8; ++v)
                pls[(v + 8 * half) * 32 + g * 16 + l16] = s01[g][v];
        __syncthreads();
        v16h pa;
        #pragma unroll
        for (int e = 0; e < 16; ++e) {
            const int kk = e + 8 * ((e >> 3) + half);
            pa[e] = (_Float16)pls[l16 * 32 + kk];
        }
        // O += P @ V
        const size_t jb = (size_t)(j0 >> 5);
        #pragma unroll
        for (int t = 0; t < 4; ++t) {
            const v16h vf = fV[(vbase + jb * 4 + t) * 32 + lane];
            o[t] = __builtin_amdgcn_wmma_f32_16x16x32_f16(
                false, pa, false, vf, (short)0, o[t], false, false);
        }
    }
    #pragma unroll
    for (int t = 0; t < 4; ++t)
        #pragma unroll
        for (int v = 0; v < 8; ++v) {
            const int irow = i0 + v + 8 * half;
            const float denom = (lrow[v] > 0.f) ? lrow[v] : 1.f;
            Ob[((size_t)(bb * S + irow)) * D + hh * HD + t * 16 + l16] = o[t][v] / denom;
        }
}

// ---------------------------------------------------------------------------
// 5) SwiGLU gate: t1 = silu(t1) * t3
// ---------------------------------------------------------------------------
__global__ void silu_mul_kernel(float* __restrict__ t1, const float* __restrict__ t3,
                                size_t n)
{
    const size_t i = (size_t)blockIdx.x * blockDim.x + threadIdx.x;
    if (i < n) {
        const float a = t1[i];
        t1[i] = (a / (1.f + __expf(-a))) * t3[i];
    }
}

// ---------------------------------------------------------------------------
// 6) boundary-predictor logits: lp = log_sigmoid(t @ bp_w2 + b2); lp[:,0] = 0
// ---------------------------------------------------------------------------
__global__ void bp_logits_kernel(const float* __restrict__ t, const float* __restrict__ w2,
                                 const float* __restrict__ b2, float* __restrict__ lp)
{
    using namespace cfg;
    __shared__ float red[256];
    const int row = blockIdx.x;   // 0..B*S-1
    float ss = 0.f;
    for (int c = threadIdx.x; c < FF; c += 256)
        ss += t[(size_t)row * FF + c] * w2[c];
    red[threadIdx.x] = ss; __syncthreads();
    for (int off = 128; off > 0; off >>= 1) {
        if (threadIdx.x < off) red[threadIdx.x] += red[threadIdx.x + off];
        __syncthreads();
    }
    if (threadIdx.x == 0) {
        const float x = red[0] + b2[0];
        const float v = (x < 0.f) ? (x - log1pf(__expf(x))) : (-log1pf(__expf(-x)));
        lp[row] = ((row % S) == 0) ? 0.f : v;
    }
}

// ---------------------------------------------------------------------------
// 7) top-K (K=256) by value (desc, ties index asc), then sort positions asc.
// ---------------------------------------------------------------------------
__global__ __launch_bounds__(1024)
void topk_kernel(const float* __restrict__ lp, int* __restrict__ pos)
{
    using namespace cfg;
    __shared__ float vals[S];
    __shared__ int   idxs[S];
    __shared__ int   p2[KQ];
    const int b = blockIdx.x;
    for (int i = threadIdx.x; i < S; i += 1024) { vals[i] = lp[b * S + i]; idxs[i] = i; }
    __syncthreads();
    for (int k = 2; k <= S; k <<= 1)
        for (int j = k >> 1; j > 0; j >>= 1) {
            for (int i = threadIdx.x; i < S; i += 1024) {
                const int ixj = i ^ j;
                if (ixj > i) {
                    const bool up = ((i & k) == 0);
                    const float va = vals[i], vb = vals[ixj];
                    const int   ia = idxs[i], ib = idxs[ixj];
                    const bool before_ixj = (vb > va) || (vb == va && ib < ia);
                    if (up ? before_ixj : !before_ixj) {
                        vals[i] = vb; vals[ixj] = va;
                        idxs[i] = ib; idxs[ixj] = ia;
                    }
                }
            }
            __syncthreads();
        }
    if (threadIdx.x < KQ) p2[threadIdx.x] = idxs[threadIdx.x];
    __syncthreads();
    for (int k = 2; k <= KQ; k <<= 1)
        for (int j = k >> 1; j > 0; j >>= 1) {
            if (threadIdx.x < KQ) {
                const int i = threadIdx.x, ixj = i ^ j;
                if (ixj > i) {
                    const bool up = ((i & k) == 0);
                    const int a = p2[i], c = p2[ixj];
                    if (up ? (c < a) : (c > a)) { p2[i] = c; p2[ixj] = a; }
                }
            }
            __syncthreads();
        }
    if (threadIdx.x < KQ) pos[b * KQ + threadIdx.x] = p2[threadIdx.x];
}

// ---------------------------------------------------------------------------
// 8) gather boundary-byte states: pe[b,k,:] = h[b, pos[b,k], :]
// ---------------------------------------------------------------------------
__global__ void gather_pe_kernel(const float* __restrict__ h, const int* __restrict__ pos,
                                 float* __restrict__ pe)
{
    using namespace cfg;
    const int bk = blockIdx.x;          // 0..B*KQ-1
    const int b  = bk / KQ;
    const int p  = pos[bk];
    for (int d = threadIdx.x; d < D; d += blockDim.x)
        pe[(size_t)bk * D + d] = h[((size_t)(b * S + p)) * D + d];
}

// ---------------------------------------------------------------------------
// 9) per-patch cross attention: query q attends to bytes [pos[p], pos[p+1])
// ---------------------------------------------------------------------------
__global__ __launch_bounds__(128)
void cross_attn_kernel(const float* __restrict__ Qb, const float* __restrict__ Kb,
                       const float* __restrict__ Vb, const int* __restrict__ pos,
                       float* __restrict__ Ob)
{
    using namespace cfg;
    __shared__ float sc[S];
    __shared__ float red[128];
    const int q  = blockIdx.x;          // 0..KQ*BKk-1
    const int hh = blockIdx.y;
    const int b  = blockIdx.z;
    const int patch = q / BKk;
    const int start = pos[b * KQ + patch];
    const int end   = (patch + 1 < KQ) ? pos[b * KQ + patch + 1] : S;
    const int len   = end - start;
    const float* qv = Qb + ((size_t)(b * KQ * BKk + q)) * D + hh * HD;

    float pm = -1e30f;
    for (int t = threadIdx.x; t < len; t += 128) {
        const float* kv = Kb + ((size_t)(b * S + start + t)) * D + hh * HD;
        float d = 0.f;
        for (int e = 0; e < HD; ++e) d += qv[e] * kv[e];
        d *= SCALE;
        sc[t] = d;
        pm = fmaxf(pm, d);
    }
    red[threadIdx.x] = pm; __syncthreads();
    for (int off = 64; off > 0; off >>= 1) {
        if (threadIdx.x < off) red[threadIdx.x] = fmaxf(red[threadIdx.x], red[threadIdx.x + off]);
        __syncthreads();
    }
    const float mx = red[0];
    __syncthreads();
    float psum = 0.f;
    for (int t = threadIdx.x; t < len; t += 128) {
        const float p = __expf(sc[t] - mx);
        sc[t] = p;
        psum += p;
    }
    red[threadIdx.x] = psum; __syncthreads();
    for (int off = 64; off > 0; off >>= 1) {
        if (threadIdx.x < off) red[threadIdx.x] += red[threadIdx.x + off];
        __syncthreads();
    }
    const float inv = 1.f / red[0];
    if (threadIdx.x < HD) {
        const int d = threadIdx.x;
        float acc = 0.f;
        for (int t = 0; t < len; ++t)
            acc += sc[t] * Vb[((size_t)(b * S + start + t)) * D + hh * HD + d];
        Ob[((size_t)(b * KQ * BKk + q)) * D + hh * HD + d] = acc * inv;
    }
}

// ---------------------------------------------------------------------------
// host-side orchestration
// ---------------------------------------------------------------------------
extern "C" void kernel_launch(void* const* d_in, const int* in_sizes, int n_in,
                              void* d_out, int out_size, void* d_ws, size_t ws_size,
                              hipStream_t stream)
{
    using namespace cfg;
    (void)in_sizes; (void)n_in; (void)out_size; (void)ws_size;

    const int*   tokens      = (const int*)  d_in[0];
    // d_in[1] = boundary_threshold (== KQ, compile-time constant here)
    const float* embedding   = (const float*)d_in[2];
    const float* hash_tables = (const float*)d_in[3];
    const float* attn_norm   = (const float*)d_in[4];
    const float* wq          = (const float*)d_in[5];
    const float* wk          = (const float*)d_in[6];
    const float* wv          = (const float*)d_in[7];
    const float* wo          = (const float*)d_in[8];
    const float* mlp_norm    = (const float*)d_in[9];
    const float* w1          = (const float*)d_in[10];
    const float* w3          = (const float*)d_in[11];
    const float* w2          = (const float*)d_in[12];
    const float* post_block_norm = (const float*)d_in[13];
    const float* bp_w1       = (const float*)d_in[14];
    const float* bp_b1       = (const float*)d_in[15];
    const float* bp_w2       = (const float*)d_in[16];
    const float* bp_b2       = (const float*)d_in[17];
    const float* q_norm_s    = (const float*)d_in[18];
    const float* kv_norm_s   = (const float*)d_in[19];
    const float* ca_wq       = (const float*)d_in[20];
    const float* ca_wk       = (const float*)d_in[21];
    const float* ca_wv       = (const float*)d_in[22];
    const float* ca_wo       = (const float*)d_in[23];
    const float* patch_proj  = (const float*)d_in[24];
    const float* out_w       = (const float*)d_in[25];
    const float* out_b       = (const float*)d_in[26];
    const float* post_pool_norm = (const float*)d_in[27];
    float* out = (float*)d_out;

    // workspace carving (256B aligned blocks)
    char* wsb = (char*)d_ws;
    size_t off = 0;
    auto alloc_bytes = [&](size_t bytes) -> void* {
        void* p = (void*)(wsb + off);
        off += (bytes + 255) & ~(size_t)255;
        return p;
    };
    auto allocf = [&](size_t n) -> float* { return (float*)alloc_bytes(n * sizeof(float)); };
    auto alloch = [&](size_t n) -> _Float16* { return (_Float16*)alloc_bytes(n * sizeof(_Float16)); };

    float* h   = allocf((size_t)BS * D);
    float* xn  = allocf((size_t)BS * D);   // norms / kvn
    float* qb  = allocf((size_t)BS * D);   // q / cross-attn Q
    float* kb  = allocf((size_t)BS * D);   // k / cross-attn K
    float* vb  = allocf((size_t)BS * D);   // v / cross-attn V
    float* ob  = allocf((size_t)BS * D);   // attention output
    float* t1  = allocf((size_t)BS * FF);  // MLP hidden / bp hidden
    float* t3  = allocf((size_t)BS * FF);
    float* lp  = allocf((size_t)BS);
    int*   pos = (int*)alloc_bytes((size_t)B * KQ * sizeof(int));
    _Float16* pA16 = alloch((size_t)BS * FF);          // packed A (max MxK)
    _Float16* pB16 = alloch((size_t)(BKk * D) * DG);   // packed B (max KxN)
    _Float16* pK16 = alloch((size_t)B * H * S * HD);   // packed attention K
    _Float16* pV16 = alloch((size_t)B * H * S * HD);   // packed attention V
    // small patch-stage buffers reuse t3 (free after SwiGLU gate)
    float* pe   = t3;
    float* qx   = pe + (size_t)B * KQ * D;
    float* qn   = qx + (size_t)B * KQ * BKk * D;
    float* co   = qn + (size_t)B * KQ * BKk * D;
    float* oc   = co + (size_t)B * KQ * BKk * D;
    float* outb = oc + (size_t)B * KQ * BKk * D;

    auto gemm = [&](const float* A, const float* Bw, const float* bias, float* C,
                    int M, int N, int K, int flags) {
        pack_a_kernel<<<dim3(M / 16, K / 32), dim3(32), 0, stream>>>(A, pA16, M, K);
        pack_b_kernel<<<dim3(N / 16, K / 32), dim3(32), 0, stream>>>(Bw, pB16, K, N);
        wmma_gemm_packed_kernel<<<dim3(M / 32, N / 64), dim3(32), 0, stream>>>(
            pA16, pB16, bias, C, M, N, K, flags);
    };

    // --- embedding + hash embeddings -------------------------------------
    embed_hash_kernel<<<BS, 256, 0, stream>>>(tokens, embedding, hash_tables, h);

    // --- local transformer blocks (L = 1) --------------------------------
    for (int l = 0; l < 1; ++l) {
        const size_t wDD = (size_t)l * D * D;
        const size_t wDF = (size_t)l * D * FF;
        const size_t wFD = (size_t)l * FF * D;
        rmsnorm_kernel<<<BS, 256, 0, stream>>>(h, attn_norm + (size_t)l * D, xn, D);
        gemm(xn, wq + wDD, nullptr, qb, BS, D, D, 0);
        gemm(xn, wk + wDD, nullptr, kb, BS, D, D, 0);
        gemm(xn, wv + wDD, nullptr, vb, BS, D, D, 0);
        pack_attn_k_kernel<<<dim3(S / 16, H, B), dim3(32), 0, stream>>>(kb, pK16);
        pack_attn_v_kernel<<<dim3(S / 32, H, B), dim3(32), 0, stream>>>(vb, pV16);
        attn_sw_kernel<<<dim3(S / 16, H, B), dim3(32), 0, stream>>>(qb, pK16, pV16, ob);
        gemm(ob, wo + wDD, nullptr, h, BS, D, D, 1);              // h += o @ wo
        rmsnorm_kernel<<<BS, 256, 0, stream>>>(h, mlp_norm + (size_t)l * D, xn, D);
        gemm(xn, w1 + wDF, nullptr, t1, BS, FF, D, 0);
        gemm(xn, w3 + wDF, nullptr, t3, BS, FF, D, 0);
        {
            const size_t n = (size_t)BS * FF;
            silu_mul_kernel<<<(unsigned)((n + 255) / 256), 256, 0, stream>>>(t1, t3, n);
        }
        gemm(t1, w2 + wFD, nullptr, h, BS, D, FF, 1);             // h += g @ w2
    }
    rmsnorm_kernel<<<BS, 256, 0, stream>>>(h, post_block_norm, h, D);   // in-place

    // --- boundary predictor + top-K patch positions ----------------------
    gemm(h, bp_w1, bp_b1, t1, BS, FF, D, 2);                      // silu(h@w1+b1)
    bp_logits_kernel<<<BS, 256, 0, stream>>>(t1, bp_w2, bp_b2, lp);
    topk_kernel<<<B, 1024, 0, stream>>>(lp, pos);

    // --- patch queries ----------------------------------------------------
    gather_pe_kernel<<<B * KQ, 256, 0, stream>>>(h, pos, pe);
    gemm(pe, patch_proj, nullptr, qx, B * KQ, D * BKk, D, 0);     // (B*K, 2D) == (B*K*2, D)

    // --- cross attention --------------------------------------------------
    rmsnorm_kernel<<<B * KQ * BKk, 256, 0, stream>>>(qx, q_norm_s, qn, D);
    rmsnorm_kernel<<<BS, 256, 0, stream>>>(h, kv_norm_s, xn, D);
    gemm(qn, ca_wq, nullptr, qb, B * KQ * BKk, D, D, 0);
    gemm(xn, ca_wk, nullptr, kb, BS, D, D, 0);
    gemm(xn, ca_wv, nullptr, vb, BS, D, D, 0);
    cross_attn_kernel<<<dim3(KQ * BKk, H, B), dim3(128), 0, stream>>>(qb, kb, vb, pos, co);
    gemm(co, ca_wo, nullptr, oc, B * KQ * BKk, D, D, 0);

    // --- out projection + post-pool norm ---------------------------------
    gemm(oc, out_w, out_b, outb, B * KQ, DG, BKk * D, 0);         // (B*K, 1536)@(1536, 2048)
    rmsnorm_kernel<<<B * KQ, 256, 0, stream>>>(outb, post_pool_norm, out, DG);
}
